// Baseline_Model_88553635709554
// MI455X (gfx1250) — matrix operand order repaired and checked by reading, not compiled, not run on hardware
//
#include <hip/hip_runtime.h>

typedef __attribute__((ext_vector_type(16))) __bf16 v16bf;
typedef __attribute__((ext_vector_type(8)))  float  v8f;

#define B_  128
#define T_  1024
#define H_  200
#define KP  224      // K padded to 7*32 for bf16 WMMA
#define NT0 13       // hidden-dim tiles of 16 (208)
#define KT  7
#define NWG 39       // 13 layer0 WGs + 26 layer1 WGs
#define HB  (B_*KP)  // bf16 state elems per parity
#define HF  (B_*H_)  // f32  state elems per parity

// ws byte offsets (all 256-aligned)
#define OFF_W    0u
#define OFF_H0B  838912u
#define OFF_H1B  953600u
#define OFF_H0F  1068288u
#define OFF_H1F  1273088u
#define OFF_BAR  1477888u
#define ZERO_DW  159808u   // dwords zeroed from OFF_H0B to end

__device__ __forceinline__ unsigned short f32_to_bf16(float f) {
  unsigned int u = __float_as_uint(f);
  u += 0x7fffu + ((u >> 16) & 1u);           // round-to-nearest-even
  return (unsigned short)(u >> 16);
}

__device__ __forceinline__ float sigmf(float v) { return 1.0f / (1.0f + __expf(-v)); }

__device__ __forceinline__ v8f wmma_bf16(v16bf a, v16bf b, v8f c) {
  // emits v_wmma_f32_16x16x32_bf16
  return __builtin_amdgcn_wmma_f32_16x16x32_bf16(false, a, false, b, (short)0, c, false, false);
}

union F8      { v8f v; float f[8]; };
union BF16x16 { uint4 q[2]; v16bf v; };

// B-fragment (K=32 x N=16 bf16): lane L -> N = L&15, K = (L>>4)*16 + half.
// Stored contiguously: 32 lanes * 16 halves; per-lane 32B -> two ds_load_b128.
__device__ __forceinline__ v16bf load_bfrag_lds(const unsigned int* sw, int frag, int lane) {
  BF16x16 u;
  const uint4* p = (const uint4*)(sw + frag * 256 + lane * 8);
  u.q[0] = p[0];
  u.q[1] = p[1];
  return u.v;
}

// A-fragment (M=16 x K=32 bf16) in contiguous K order: lane L -> M = L&15, g = L>>4;
// halves 0..15 cover K = g*16 + {0..15} (matches B-fragment packing).
// K-pairs are contiguous -> two global_load_b128 per fragment.
__device__ __forceinline__ v16bf load_afrag(const unsigned short* hb, int mbase, int kbase, int lane) {
  const int mr = mbase + (lane & 15);
  const int g  = lane >> 4;
  const unsigned int* r = (const unsigned int*)(hb + mr * KP + kbase);
  BF16x16 u;
  u.q[0] = *((const uint4*)r + 2 * g);        // dwords 8g..8g+3  -> K = 16g..16g+7
  u.q[1] = *((const uint4*)r + 2 * g + 1);  // dwords 8g+4..    -> K = 16g+8..16g+15
  return u.v;
}

__global__ void zero_ws_kernel(unsigned int* p, unsigned int n) {
  for (unsigned int i = blockIdx.x * 256 + threadIdx.x; i < n; i += gridDim.x * 256) p[i] = 0u;
}

// Pack W (600x200 row-major f32) into bf16 B-fragments:
// frag linear index = ((mat*3+gate)*13 + nt)*7 + kt ; element = lane*16 + half.
// B[k][n] = W[gate*200 + n][k], zero-padded to N=208, K=224.
__global__ void pack_weights_kernel(const float* __restrict__ Whh0,
                                    const float* __restrict__ Wih1,
                                    const float* __restrict__ Whh1,
                                    unsigned short* __restrict__ pw) {
  const int idx = blockIdx.x * 256 + threadIdx.x;
  if (idx >= 3 * 3 * NT0 * KT * 512) return;
  const int half = idx & 15;
  const int lane = (idx >> 4) & 31;
  int frag = idx >> 9;
  const int kt = frag % KT;  frag /= KT;
  const int nt = frag % NT0; frag /= NT0;
  const int gate = frag % 3;
  const int mat  = frag / 3;
  const int n = nt * 16 + (lane & 15);
  const int k = kt * 32 + (lane >> 4) * 16 + half;
  const float* W = (mat == 0) ? Whh0 : ((mat == 1) ? Wih1 : Whh1);
  const float v = (n < H_ && k < H_) ? W[(gate * H_ + n) * H_ + k] : 0.0f;
  pw[idx] = f32_to_bf16(v);
}

__device__ __forceinline__ void grid_barrier(unsigned int* bar) {
  __syncthreads();
  if (threadIdx.x == 0) {
    __threadfence();
    volatile unsigned int* vgen = bar + 1;
    const unsigned int g = *vgen;
    if (atomicAdd(bar, 1u) == NWG - 1u) {
      *(volatile unsigned int*)bar = 0u;
      __threadfence();
      atomicAdd(bar + 1, 1u);
    } else {
      while (*vgen == g) __builtin_amdgcn_s_sleep(2);
    }
    __threadfence();
  }
  __syncthreads();
}

__global__ __launch_bounds__(128, 1) void gru_persistent_kernel(
    const float* __restrict__ x,
    const float* __restrict__ Wih0, const float* __restrict__ bih0, const float* __restrict__ bhh0,
    const float* __restrict__ bih1, const float* __restrict__ bhh1,
    const float* __restrict__ Wfc,  const float* __restrict__ bfc,
    const unsigned short* __restrict__ pw,
    unsigned short* __restrict__ h0b, unsigned short* __restrict__ h1b,
    float* __restrict__ h0f, float* __restrict__ h1f,
    unsigned int* __restrict__ bar, float* __restrict__ out)
{
  __shared__ unsigned int smemW[10752];   // up to 42 B-fragments (1 KB each)

  const int  tid  = threadIdx.x;
  const int  lane = tid & 31;
  const int  wv   = tid >> 5;
  const bool isL0 = blockIdx.x < NT0;
  const int  nt   = isL0 ? (int)blockIdx.x : (((int)blockIdx.x - NT0) >> 1);

  // ---- stage this WG's weight slice into LDS once (stays for all 1024 steps) ----
  const unsigned int* pwd = (const unsigned int*)pw;
  const int nfrag = isL0 ? 21 : 42;
  for (int f = 0; f < nfrag; ++f) {
    int mat, gate, kt;
    if (isL0) { mat = 0; gate = f / 7; kt = f % 7; }
    else      { mat = 1 + f / 21; const int r = f % 21; gate = r / 7; kt = r % 7; }
    const int gfrag = ((mat * 3 + gate) * NT0 + nt) * KT + kt;
    for (int d = tid; d < 256; d += 128)
      smemW[f * 256 + d] = pwd[gfrag * 256 + d];
  }
  __syncthreads();

  // ---- per-lane constants (C/D layout: this lane owns column n_l for all M) ----
  const int  n_l  = nt * 16 + (lane & 15);
  const bool nval = n_l < H_;
  float wr[8], wz[8], wn[8];
  float bir = 0.f, biz = 0.f, bin_ = 0.f, bhr = 0.f, bhz = 0.f, bhn = 0.f;
  if (nval) {
    if (isL0) {
#pragma unroll
      for (int c = 0; c < 8; ++c) {
        wr[c] = Wih0[n_l * 8 + c];
        wz[c] = Wih0[(H_ + n_l) * 8 + c];
        wn[c] = Wih0[(2 * H_ + n_l) * 8 + c];
      }
      bir = bih0[n_l]; biz = bih0[H_ + n_l]; bin_ = bih0[2 * H_ + n_l];
      bhr = bhh0[n_l]; bhz = bhh0[H_ + n_l]; bhn  = bhh0[2 * H_ + n_l];
    } else {
      bir = bih1[n_l]; biz = bih1[H_ + n_l]; bin_ = bih1[2 * H_ + n_l];
      bhr = bhh1[n_l]; bhz = bhh1[H_ + n_l]; bhn  = bhh1[2 * H_ + n_l];
    }
  }
  const v8f zf = {0.f, 0.f, 0.f, 0.f, 0.f, 0.f, 0.f, 0.f};

  // Pipeline: tick tau: layer0 -> h0^(tau+1); layer1 -> h1^(tau); fc -> out_{tau-2}.
  for (int tau = 0; tau <= T_ + 1; ++tau) {
    if (isL0) {
      // ---- FC head: out_{tau-2} = tanh(h1^{(tau-1)} @ Wfc^T + bfc) ----
      if (tau >= 2) {
        const int lin = (int)blockIdx.x * 128 + tid;
        if (lin < B_ * 4) {
          const int b = lin >> 2, o = lin & 3;
          const float* hv = h1f + ((tau - 1) & 1) * HF + b * H_;
          const float* wf = Wfc + o * H_;
          float s = bfc[o];
          for (int j = 0; j < H_; ++j) s += hv[j] * wf[j];
          out[((size_t)b * T_ + (tau - 2)) * 4 + o] = tanhf(s);
        }
      }
      // ---- layer-0 GRU step (uniform control flow around WMMA) ----
      if (tau < T_) {
        const int pr = tau & 1, pwp = pr ^ 1;
        const unsigned short* hrd = h0b + pr * HB;
        unsigned short*       hwr = h0b + pwp * HB;
        const float*          hfr = h0f + pr * HF;
        float*                hfw = h0f + pwp * HF;
#pragma unroll
        for (int mi = 0; mi < 2; ++mi) {
          const int mb = (wv * 2 + mi) * 16;
          F8 ar, az, an; ar.v = zf; az.v = zf; an.v = zf;
#pragma unroll
          for (int kt = 0; kt < KT; ++kt) {
            const v16bf a = load_afrag(hrd, mb, kt * 32, lane);
            ar.v = wmma_bf16(a, load_bfrag_lds(smemW,      kt, lane), ar.v);
            az.v = wmma_bf16(a, load_bfrag_lds(smemW,  7 + kt, lane), az.v);
            an.v = wmma_bf16(a, load_bfrag_lds(smemW, 14 + kt, lane), an.v);
          }
          if (nval) {
#pragma unroll
            for (int v = 0; v < 8; ++v) {
              const int m = mb + v + 8 * (lane >> 4);
              float4 tf4 = make_float4(1.f, 1.f, 1.f, 1.f);
              if (tau > 0) tf4 = *(const float4*)(x + ((size_t)m * T_ + (tau - 1)) * 8);
              const float4 em = *(const float4*)(x + (size_t)m * T_ * 8 + 4);
              float ixr = bir + wr[0]*tf4.x + wr[1]*tf4.y + wr[2]*tf4.z + wr[3]*tf4.w
                              + wr[4]*em.x  + wr[5]*em.y  + wr[6]*em.z  + wr[7]*em.w;
              float ixz = biz + wz[0]*tf4.x + wz[1]*tf4.y + wz[2]*tf4.z + wz[3]*tf4.w
                              + wz[4]*em.x  + wz[5]*em.y  + wz[6]*em.z  + wz[7]*em.w;
              float ixn = bin_ + wn[0]*tf4.x + wn[1]*tf4.y + wn[2]*tf4.z + wn[3]*tf4.w
                               + wn[4]*em.x  + wn[5]*em.y  + wn[6]*em.z  + wn[7]*em.w;
              const float rg = sigmf(ixr + ar.f[v] + bhr);
              const float zg = sigmf(ixz + az.f[v] + bhz);
              const float ng = tanhf(ixn + rg * (an.f[v] + bhn));
              const float ho = hfr[m * H_ + n_l];
              const float hv2 = (1.f - zg) * ng + zg * ho;
              hfw[m * H_ + n_l] = hv2;
              hwr[m * KP + n_l] = f32_to_bf16(hv2);
            }
          }
        }
      }
    } else if (tau >= 1 && tau <= T_) {
      // ---- layer-1 GRU step: gx1 = h0_new @ Wih1^T ; gh1 = h1 @ Whh1^T ----
      const int idx1 = (int)blockIdx.x - NT0;
      const int mb = ((idx1 & 1) * 4 + wv) * 16;
      const unsigned short* h0rd = h0b + (tau & 1) * HB;
      const unsigned short* h1rd = h1b + ((tau - 1) & 1) * HB;
      unsigned short*       h1wr = h1b + (tau & 1) * HB;
      const float*          h1fr = h1f + ((tau - 1) & 1) * HF;
      float*                h1fw = h1f + (tau & 1) * HF;
      F8 xr, xz, xn, hr, hz, hn;
      xr.v = zf; xz.v = zf; xn.v = zf; hr.v = zf; hz.v = zf; hn.v = zf;
#pragma unroll
      for (int kt = 0; kt < KT; ++kt) {
        const v16bf a0 = load_afrag(h0rd, mb, kt * 32, lane);
        xr.v = wmma_bf16(a0, load_bfrag_lds(smemW,      kt, lane), xr.v);
        xz.v = wmma_bf16(a0, load_bfrag_lds(smemW,  7 + kt, lane), xz.v);
        xn.v = wmma_bf16(a0, load_bfrag_lds(smemW, 14 + kt, lane), xn.v);
        const v16bf a1 = load_afrag(h1rd, mb, kt * 32, lane);
        hr.v = wmma_bf16(a1, load_bfrag_lds(smemW, 21 + kt, lane), hr.v);
        hz.v = wmma_bf16(a1, load_bfrag_lds(smemW, 28 + kt, lane), hz.v);
        hn.v = wmma_bf16(a1, load_bfrag_lds(smemW, 35 + kt, lane), hn.v);
      }
      if (nval) {
#pragma unroll
        for (int v = 0; v < 8; ++v) {
          const int m = mb + v + 8 * (lane >> 4);
          const float rg = sigmf(xr.f[v] + bir + hr.f[v] + bhr);
          const float zg = sigmf(xz.f[v] + biz + hz.f[v] + bhz);
          const float ng = tanhf(xn.f[v] + bin_ + rg * (hn.f[v] + bhn));
          const float ho = h1fr[m * H_ + n_l];
          const float hv2 = (1.f - zg) * ng + zg * ho;
          h1fw[m * H_ + n_l] = hv2;
          h1wr[m * KP + n_l] = f32_to_bf16(hv2);
        }
      }
    }
    if (tau <= T_) grid_barrier(bar);
  }
}

extern "C" void kernel_launch(void* const* d_in, const int* in_sizes, int n_in,
                              void* d_out, int out_size, void* d_ws, size_t ws_size,
                              hipStream_t stream) {
  (void)in_sizes; (void)n_in; (void)out_size; (void)ws_size;
  const float* x    = (const float*)d_in[0];
  const float* Wih0 = (const float*)d_in[1];
  const float* Whh0 = (const float*)d_in[2];
  const float* bih0 = (const float*)d_in[3];
  const float* bhh0 = (const float*)d_in[4];
  const float* Wih1 = (const float*)d_in[5];
  const float* Whh1 = (const float*)d_in[6];
  const float* bih1 = (const float*)d_in[7];
  const float* bhh1 = (const float*)d_in[8];
  const float* Wfc  = (const float*)d_in[9];
  const float* bfc  = (const float*)d_in[10];

  char* ws = (char*)d_ws;
  unsigned short* pw  = (unsigned short*)(ws + OFF_W);
  unsigned short* h0b = (unsigned short*)(ws + OFF_H0B);
  unsigned short* h1b = (unsigned short*)(ws + OFF_H1B);
  float* h0f = (float*)(ws + OFF_H0F);
  float* h1f = (float*)(ws + OFF_H1F);
  unsigned int* bar = (unsigned int*)(ws + OFF_BAR);
  float* out = (float*)d_out;

  // 1) zero state buffers + barrier (deterministic per call, graph-capture safe)
  zero_ws_kernel<<<625, 256, 0, stream>>>((unsigned int*)(ws + OFF_H0B), ZERO_DW);
  // 2) pack recurrent weights into wave32 WMMA B-fragment layout (bf16)
  pack_weights_kernel<<<1638, 256, 0, stream>>>(Whh0, Wih1, Whh1, pw);
  // 3) persistent pipelined GRU: 39 WGs, one grid barrier per timestep
  gru_persistent_kernel<<<NWG, 128, 0, stream>>>(x, Wih0, bih0, bhh0, bih1, bhh1,
                                                 Wfc, bfc, pw, h0b, h1b, h0f, h1f,
                                                 bar, out);
}
